// GRImplicitNet_91311004713112
// MI455X (gfx1250) — matrix operand order repaired and checked by reading, not compiled, not run on hardware
//
#include <hip/hip_runtime.h>
#include <math.h>

// GRNet-style implicit net forward for MI455X (gfx1250).
// All matmul-shaped work (conv im2col-GEMM, parity-decomposed deconv GEMM,
// FC, MLP) runs through v_wmma_f32_16x16x32_f16 (f16 in, f32 accumulate).
// Activations/weights are fp16 (workload is bandwidth-bound: ~100 GFLOP vs
// several hundred MB of activation traffic at 23.3 TB/s HBM; most of the
// working set is L2-resident at 192 MB). GEMM staging uses 16-byte
// global_load_b128 only (A: ds_store_b128, B: 2-byte LDS stores) (call sites guarantee K%8==0 and
// N%8==0 so 8-half segments are all-or-nothing in bounds), a transposed
// B tile in LDS so fragments load as ds_load_b128, and double-buffered LDS
// with one barrier per K step. Workspace ~450 MB, bump-allocated from d_ws.

typedef _Float16 h16;
typedef __attribute__((ext_vector_type(16))) _Float16 v16h;
typedef __attribute__((ext_vector_type(8)))  _Float16 v8h;
typedef __attribute__((ext_vector_type(8)))  float    v8f;

__device__ __forceinline__ float act_apply(float v, int act) {
  if (act == 1) return v > 0.f ? v : 0.2f  * v;   // lrelu 0.2
  if (act == 2) return v > 0.f ? v : 0.01f * v;   // lrelu 0.01
  if (act == 3) return 1.f / (1.f + __expf(-v));  // sigmoid
  return v;
}

// ---------------------------------------------------------------------------
// WMMA GEMM: C = act(A[M,K] * B[K,N] + bias[M]).
// Contract: K % 8 == 0, N % 8 == 0, A/B 16-byte aligned.
// Block tile 64x64, BK=32. 256 threads = 8 wave32 waves arranged 2(M) x 4(N);
// each wave owns 32x16 of C via two v_wmma_f32_16x16x32_f16 accumulators.
// Stores only columns < nstore (live columns of chunked outputs).
// ---------------------------------------------------------------------------
__global__ __launch_bounds__(256) void k_gemm(
    const h16* __restrict__ A, const h16* __restrict__ B,
    const float* __restrict__ bias,
    float* __restrict__ Cf, h16* __restrict__ Ch,
    int M, int N, int K, int ldc, int nstore, int act)
{
  __shared__ h16 As[2][64][40];   // row-major A tile, padded rows (80 B)
  __shared__ h16 Bt[2][64][40];   // TRANSPOSED B tile: Bt[col][k]

  const int tid  = threadIdx.x;
  const int lane = tid & 31;
  const int wave = tid >> 5;
  const int wm   = wave >> 2;        // 0..1 : 32-row slab
  const int wn   = wave & 3;         // 0..3 : 16-col slab
  const int bm   = blockIdx.y * 64;
  const int bn   = blockIdx.x * 64;
  const int l    = lane & 15;
  const int hi   = lane >> 4;

  // staging: one 16B segment per thread per tile side
  const int ar = tid >> 2, ac = (tid & 3) * 8;   // A: 64 rows x 32 cols
  const int br = tid >> 3, bc = (tid & 7) * 8;   // B: 32 k-rows x 64 cols
  const int gr  = bm + ar;
  const int gcb = bn + bc;

  v8f acc0 = {0.f,0.f,0.f,0.f,0.f,0.f,0.f,0.f};
  v8f acc1 = acc0;

  auto loadA = [&](int k0) -> uint4 {
    uint4 v = {0u, 0u, 0u, 0u};
    if (gr < M && (k0 + ac + 8) <= K)          // all-or-nothing (K%8==0, ac%8==0)
      v = *(const uint4*)(A + (size_t)gr * K + k0 + ac);
    return v;
  };
  auto loadB = [&](int k0) -> uint4 {
    uint4 v = {0u, 0u, 0u, 0u};
    int grb = k0 + br;
    if (grb < K && (gcb + 8) <= N)             // all-or-nothing (N%8==0, bc%8==0)
      v = *(const uint4*)(B + (size_t)grb * N + gcb);
    return v;
  };
  auto stageA = [&](int buf, uint4 v) { *(uint4*)&As[buf][ar][ac] = v; };
  auto stageB = [&](int buf, uint4 v) {
    h16 hv[8];
    __builtin_memcpy(hv, &v, 16);
#pragma unroll
    for (int j = 0; j < 8; ++j) Bt[buf][bc + j][br] = hv[j];
  };

  // prologue: stage tile 0
  {
    uint4 av = loadA(0), bv = loadB(0);
    stageA(0, av);
    stageB(0, bv);
  }
  __syncthreads();

  int buf = 0;
  for (int k0 = 0; k0 < K; k0 += 32) {
    const bool more = (k0 + 32) < K;
    uint4 av2 = {0u,0u,0u,0u}, bv2 = {0u,0u,0u,0u};
    if (more) {
      av2 = loadA(k0 + 32);
      bv2 = loadB(k0 + 32);
      if (k0 + 64 < K) {  // speculative prefetch two tiles ahead
        if (gr < M) __builtin_prefetch(&A[(size_t)gr * K + k0 + 64 + ac], 0, 1);
        if (gcb + 8 <= N) __builtin_prefetch(&B[(size_t)(k0 + 64 + br) * N + gcb], 0, 1);
      }
    }

    // fragments (all contiguous, 16B-aligned LDS reads -> ds_load_b128)
    const h16* Ar0 = &As[buf][wm * 32 + l][0];
    const h16* Ar1 = &As[buf][wm * 32 + 16 + l][0];
    const h16* Brw = &Bt[buf][wn * 16 + l][0];
    v8h a0l = *(const v8h*)(Ar0 + hi * 8);
    v8h a0h = *(const v8h*)(Ar0 + 16 + hi * 8);
    v8h a1l = *(const v8h*)(Ar1 + hi * 8);
    v8h a1h = *(const v8h*)(Ar1 + 16 + hi * 8);
    v8h bl  = *(const v8h*)(Brw + hi * 8);
    v8h bh  = *(const v8h*)(Brw + 16 + hi * 8);
    v16h a0 = __builtin_shufflevector(a0l, a0h, 0,1,2,3,4,5,6,7,8,9,10,11,12,13,14,15);
    v16h a1 = __builtin_shufflevector(a1l, a1h, 0,1,2,3,4,5,6,7,8,9,10,11,12,13,14,15);
    v16h bf = __builtin_shufflevector(bl,  bh,  0,1,2,3,4,5,6,7,8,9,10,11,12,13,14,15);

    acc0 = __builtin_amdgcn_wmma_f32_16x16x32_f16(false, a0, false, bf, (short)0, acc0, false, false);
    acc1 = __builtin_amdgcn_wmma_f32_16x16x32_f16(false, a1, false, bf, (short)0, acc1, false, false);

    if (more) {  // stage next tile into the other buffer
      stageA(buf ^ 1, av2);
      stageB(buf ^ 1, bv2);
    }
    __syncthreads();
    buf ^= 1;
  }

  const int col = bn + wn * 16 + l;
  if (col < nstore) {
#pragma unroll
    for (int r = 0; r < 8; ++r) {
      int row0 = bm + wm * 32 + hi * 8 + r;
      int row1 = row0 + 16;
      if (row0 < M) {
        float v = acc0[r] + (bias ? bias[row0] : 0.f);
        v = act_apply(v, act);
        if (Cf) Cf[(size_t)row0 * ldc + col] = v;
        if (Ch) Ch[(size_t)row0 * ldc + col] = (h16)v;
      }
      if (row1 < M) {
        float v = acc1[r] + (bias ? bias[row1] : 0.f);
        v = act_apply(v, act);
        if (Cf) Cf[(size_t)row1 * ldc + col] = v;
        if (Ch) Ch[(size_t)row1 * ldc + col] = (h16)v;
      }
    }
  }
}

// ---------------------------------------------------------------------------
// Elementwise / data movement kernels
// ---------------------------------------------------------------------------
__global__ void k_zero_f32(float* __restrict__ p, long n) {
  for (long i = blockIdx.x * (long)blockDim.x + threadIdx.x; i < n;
       i += (long)gridDim.x * blockDim.x) p[i] = 0.f;
}

__global__ void k_f32_to_f16(const float* __restrict__ s, h16* __restrict__ d, long n) {
  for (long i = blockIdx.x * (long)blockDim.x + threadIdx.x; i < n;
       i += (long)gridDim.x * blockDim.x) d[i] = (h16)s[i];
}

// Trilinear scatter of unit mass into [B][64^3] (atomics).
__global__ void k_gridding(const float* __restrict__ pts, float* __restrict__ grid,
                           int Bn, int Np) {
  int i = blockIdx.x * blockDim.x + threadIdx.x;
  if (i >= Bn * Np) return;
  int b = i / Np;
  const float* p = pts + (size_t)i * 3;
  float gx = p[0] * 32.f + 32.f, gy = p[1] * 32.f + 32.f, gz = p[2] * 32.f + 32.f;
  float fx = gx - floorf(gx), fy = gy - floorf(gy), fz = gz - floorf(gz);
  int lx = (int)floorf(gx), ly = (int)floorf(gy), lz = (int)floorf(gz);
  lx = lx < 0 ? 0 : (lx > 62 ? 62 : lx);
  ly = ly < 0 ? 0 : (ly > 62 ? 62 : ly);
  lz = lz < 0 ? 0 : (lz > 62 ? 62 : lz);
  float* g = grid + (size_t)b * 262144;
#pragma unroll
  for (int dx = 0; dx < 2; ++dx)
#pragma unroll
    for (int dy = 0; dy < 2; ++dy)
#pragma unroll
      for (int dz = 0; dz < 2; ++dz) {
        float w = (dx ? fx : 1.f - fx) * (dy ? fy : 1.f - fy) * (dz ? fz : 1.f - fz);
        atomicAdd(&g[(lx + dx) * 4096 + (ly + dy) * 64 + (lz + dz)], w);
      }
}

// im2col for conv3d(k=4,p=2): x [Ci][B][D^3] -> col [K=Ci*64][ncPad].
// Columns beyond Nfull (global) are zero-filled (ncPad is 8-aligned).
__global__ void k_im2col_conv(const h16* __restrict__ x, h16* __restrict__ col,
                              int Ci, int Bn, int D, int O, long col0, int ncPad,
                              long Nfull) {
  const long total = (long)Ci * 64 * ncPad;
  const int D2 = D * D, D3 = D2 * D;
  for (long i = blockIdx.x * (long)blockDim.x + threadIdx.x; i < total;
       i += (long)gridDim.x * blockDim.x) {
    int n = (int)(i % ncPad);
    int k = (int)(i / ncPad);
    int tap = k & 63, ci = k >> 6;
    int kw = tap & 3, kh = (tap >> 2) & 3, kd = tap >> 4;
    long gcl = col0 + n;
    h16 v = (h16)0.f;
    if (gcl < Nfull) {
      int gc = (int)gcl;
      int ow = gc % O; int t = gc / O; int oh = t % O; t /= O; int od = t % O; int b = t / O;
      int id = od + kd - 2, ih = oh + kh - 2, iw = ow + kw - 2;
      if ((unsigned)id < (unsigned)D && (unsigned)ih < (unsigned)D && (unsigned)iw < (unsigned)D)
        v = x[((size_t)ci * Bn + b) * D3 + (size_t)id * D2 + ih * D + iw];
    }
    col[i] = v;
  }
}

// Per-channel BN stats (training mode: batch+spatial), block per channel.
__global__ void k_bn_stats(const h16* __restrict__ x, int L,
                           float* __restrict__ mean, float* __restrict__ istd, float eps) {
  int c = blockIdx.x;
  const h16* p = x + (size_t)c * L;
  float s = 0.f, s2 = 0.f;
  for (int i = threadIdx.x; i < L; i += blockDim.x) {
    float v = (float)p[i]; s += v; s2 += v * v;
  }
  __shared__ float sh[256], sh2[256];
  sh[threadIdx.x] = s; sh2[threadIdx.x] = s2;
  __syncthreads();
  for (int o = 128; o > 0; o >>= 1) {
    if (threadIdx.x < o) { sh[threadIdx.x] += sh[threadIdx.x + o]; sh2[threadIdx.x] += sh2[threadIdx.x + o]; }
    __syncthreads();
  }
  if (threadIdx.x == 0) {
    float m = sh[0] / (float)L;
    float v = sh2[0] / (float)L - m * m;
    mean[c] = m;
    istd[c] = rsqrtf(v + eps);
  }
}

// Fused BN + LeakyReLU(0.2) + 2x2x2 maxpool: [C][B][O^3] -> [C][B][P^3]
__global__ void k_bn_lrelu_pool(const h16* __restrict__ x,
                                const float* __restrict__ mean, const float* __restrict__ istd,
                                const float* __restrict__ ga, const float* __restrict__ be,
                                h16* __restrict__ y, int C, int Bn, int O, int P) {
  const long total = (long)C * Bn * P * P * P;
  const long O2 = (long)O * O, O3 = O2 * O;
  for (long i = blockIdx.x * (long)blockDim.x + threadIdx.x; i < total;
       i += (long)gridDim.x * blockDim.x) {
    int pw = (int)(i % P); long t = i / P;
    int ph = (int)(t % P); t /= P;
    int pd = (int)(t % P); t /= P;
    int b  = (int)(t % Bn); int c = (int)(t / Bn);
    const h16* base = x + ((size_t)c * Bn + b) * O3;
    float m = mean[c], is = istd[c], g = ga[c], bb = be[c];
    float best = -3.0e38f;
#pragma unroll
    for (int dd = 0; dd < 2; ++dd)
#pragma unroll
      for (int hh = 0; hh < 2; ++hh)
#pragma unroll
        for (int ww = 0; ww < 2; ++ww) {
          float v = (float)base[(size_t)(2 * pd + dd) * O2 + (2 * ph + hh) * O + (2 * pw + ww)];
          v = (v - m) * is * g + bb;
          v = v > 0.f ? v : 0.2f * v;
          best = fmaxf(best, v);
        }
    y[i] = (h16)best;
  }
}

// Pack ConvTranspose3d(k=4,s=2,p=1) weights [Ci][Co][4][4][4] (f32)
// into 8 parity-class matrices: dst[par][Co][Ci*8] (f16).
__global__ void k_pack_deconv_w(const float* __restrict__ w, h16* __restrict__ dst,
                                int Ci, int Co) {
  long total = (long)8 * Co * Ci * 8;
  for (long i = blockIdx.x * (long)blockDim.x + threadIdx.x; i < total;
       i += (long)gridDim.x * blockDim.x) {
    int t = (int)(i & 7); long r = i >> 3;
    int ci = (int)(r % Ci); r /= Ci;
    int co = (int)(r % Co); int par = (int)(r / Co);
    int tw = t & 1, th = (t >> 1) & 1, td = (t >> 2) & 1;
    int pd = par >> 2, ph = (par >> 1) & 1, pw = par & 1;
    int kd = pd ? (td ? 2 : 0) : (td ? 3 : 1);
    int kh = ph ? (th ? 2 : 0) : (th ? 3 : 1);
    int kw = pw ? (tw ? 2 : 0) : (tw ? 3 : 1);
    dst[i] = (h16)w[(((size_t)ci * Co + co) * 4 + kd) * 16 + kh * 4 + kw];
  }
}

// im2col for one deconv parity class: in [Ci][B][D^3] -> col [K=Ci*8][B*D^3]
__global__ void k_im2col_deconv(const h16* __restrict__ x, h16* __restrict__ col,
                                int Ci, int Bn, int D, int par) {
  const int D2 = D * D, D3 = D2 * D;
  const long Ncols = (long)Bn * D3;
  const long total = (long)Ci * 8 * Ncols;
  const int pd = par >> 2, ph = (par >> 1) & 1, pw = par & 1;
  for (long i = blockIdx.x * (long)blockDim.x + threadIdx.x; i < total;
       i += (long)gridDim.x * blockDim.x) {
    long n = i % Ncols;
    int k = (int)(i / Ncols);
    int t = k & 7, ci = k >> 3;
    int tw = t & 1, th = (t >> 1) & 1, td = (t >> 2) & 1;
    int qw = (int)(n % D); long r = n / D;
    int qh = (int)(r % D); r /= D;
    int qd = (int)(r % D); int b = (int)(r / D);
    int id = qd + (pd ? (td ? 0 : 1) : (td ? -1 : 0));
    int ih = qh + (ph ? (th ? 0 : 1) : (th ? -1 : 0));
    int iw = qw + (pw ? (tw ? 0 : 1) : (tw ? -1 : 0));
    h16 v = (h16)0.f;
    if ((unsigned)id < (unsigned)D && (unsigned)ih < (unsigned)D && (unsigned)iw < (unsigned)D)
      v = x[((size_t)ci * Bn + b) * D3 + (size_t)id * D2 + ih * D + iw];
    col[i] = v;
  }
}

// Scatter one parity-class GEMM output [Co][B*D^3] into full [Co][B][(2D)^3]
__global__ void k_scatter_par(const h16* __restrict__ src, h16* __restrict__ dst,
                              int Co, int Bn, int D, int par) {
  const int pd = par >> 2, ph = (par >> 1) & 1, pw = par & 1;
  const int D3 = D * D * D, O = 2 * D;
  const long O3 = (long)O * O * O;
  const long total = (long)Co * Bn * D3;
  for (long i = blockIdx.x * (long)blockDim.x + threadIdx.x; i < total;
       i += (long)gridDim.x * blockDim.x) {
    int r = (int)(i % D3); long cb = i / D3;
    int b = (int)(cb % Bn); int c = (int)(cb / Bn);
    int qw = r % D; int tt = r / D; int qh = tt % D; int qd = tt / D;
    int od = 2 * qd + pd, oh = 2 * qh + ph, ow = 2 * qw + pw;
    dst[((size_t)c * Bn + b) * O3 + (size_t)od * O * O + oh * O + ow] = src[i];
  }
}

// Decoder epilogue: y = relu(bn(x)) + skip (skip nullable). L = per-channel len.
__global__ void k_bn_relu_add(const h16* __restrict__ x,
                              const float* __restrict__ mean, const float* __restrict__ istd,
                              const float* __restrict__ ga, const float* __restrict__ be,
                              const h16* __restrict__ skip, h16* __restrict__ y,
                              int C, long L) {
  const long total = (long)C * L;
  for (long i = blockIdx.x * (long)blockDim.x + threadIdx.x; i < total;
       i += (long)gridDim.x * blockDim.x) {
    int c = (int)(i / L);
    float v = ((float)x[i] - mean[c]) * istd[c] * ga[c] + be[c];
    v = fmaxf(v, 0.f);
    if (skip) v += (float)skip[i];
    y[i] = (h16)v;
  }
}

// [256][B][64] -> [16384][8] for fc5 (torch reshape order c*64+s), zero-pad b>=Bn
__global__ void k_pack_fc(const h16* __restrict__ p4, h16* __restrict__ xfc, int Bn) {
  long total = (long)16384 * 8;
  for (long i = blockIdx.x * (long)blockDim.x + threadIdx.x; i < total;
       i += (long)gridDim.x * blockDim.x) {
    int b = (int)(i & 7); int k = (int)(i >> 3);
    int c = k >> 6, s = k & 63;
    xfc[i] = (b < Bn) ? p4[((size_t)c * Bn + b) * 64 + s] : (h16)0.f;
  }
}

// f4r = f4l + reshape(h6): f4r[c][b][s] = f4l[c][b][s] + h6[(c*64+s)][b] (ld 8)
__global__ void k_add_fc(const h16* __restrict__ p4, const h16* __restrict__ h6,
                         h16* __restrict__ f4r, int Bn) {
  long total = (long)256 * Bn * 64;
  for (long i = blockIdx.x * (long)blockDim.x + threadIdx.x; i < total;
       i += (long)gridDim.x * blockDim.x) {
    int s = (int)(i & 63); long r = i >> 6;
    int b = (int)(r % Bn); int c = (int)(r / Bn);
    f4r[i] = (h16)((float)p4[i] + (float)h6[(size_t)(c * 64 + s) * 8 + b]);
  }
}

// 8-corner gather projection: f64r [16][B][64^3] -> feat [16][B*NQ]
__global__ void k_project(const float* __restrict__ pts, const h16* __restrict__ f,
                          h16* __restrict__ feat, int Bn, int NQ) {
  const long BNQ = (long)Bn * NQ;
  const long total = 16 * BNQ;
  for (long i = blockIdx.x * (long)blockDim.x + threadIdx.x; i < total;
       i += (long)gridDim.x * blockDim.x) {
    long n = i % BNQ; int c = (int)(i / BNQ);
    int b = (int)(n / NQ); int q = (int)(n % NQ);
    const float* p = pts + ((size_t)b * NQ + q) * 3;
    int x = (int)floorf(p[0] * 32.f) + 32;
    int y = (int)floorf(p[1] * 32.f) + 32;
    int z = (int)floorf(p[2] * 32.f) + 32;
    x = x < 0 ? 0 : (x > 62 ? 62 : x);
    y = y < 0 ? 0 : (y > 62 ? 62 : y);
    z = z < 0 ? 0 : (z > 62 ? 62 : z);
    const h16* base = f + ((size_t)c * Bn + b) * 262144;
    float s = 0.f;
#pragma unroll
    for (int dx = 0; dx < 2; ++dx)
#pragma unroll
      for (int dy = 0; dy < 2; ++dy)
#pragma unroll
        for (int dz = 0; dz < 2; ++dz)
          s += (float)base[(x + dx) * 4096 + (y + dy) * 64 + (z + dz)];
    feat[i] = (h16)s;
  }
}

__global__ void k_mse(const float* __restrict__ preds, const float* __restrict__ labels,
                      float* __restrict__ err, int n) {
  float s = 0.f;
  for (int i = blockIdx.x * blockDim.x + threadIdx.x; i < n;
       i += gridDim.x * blockDim.x) {
    float d = preds[i] - labels[i];
    s += d * d;
  }
  __shared__ float sh[256];
  sh[threadIdx.x] = s;
  __syncthreads();
  for (int o = 128; o > 0; o >>= 1) {
    if (threadIdx.x < o) sh[threadIdx.x] += sh[threadIdx.x + o];
    __syncthreads();
  }
  if (threadIdx.x == 0) atomicAdd(err, sh[0] / (float)n);
}

// ---------------------------------------------------------------------------
// Host orchestration
// ---------------------------------------------------------------------------
static inline int nblk(long total) {
  long b = (total + 255) / 256;
  return (int)(b < 4096 ? (b > 0 ? b : 1) : 4096);
}

extern "C" void kernel_launch(void* const* d_in, const int* in_sizes, int n_in,
                              void* d_out, int out_size, void* d_ws, size_t ws_size,
                              hipStream_t stream) {
  (void)in_sizes; (void)n_in; (void)out_size; (void)ws_size;

  const int  Bn = 4, NP = 2048, NQ = 8192;
  const float* partial = (const float*)d_in[0];
  const float* points  = (const float*)d_in[1];
  const float* labels  = (const float*)d_in[2];
  const float* conv_w[4] = {(const float*)d_in[3],  (const float*)d_in[7],
                            (const float*)d_in[11], (const float*)d_in[15]};
  const float* conv_b[4] = {(const float*)d_in[4],  (const float*)d_in[8],
                            (const float*)d_in[12], (const float*)d_in[16]};
  const float* ebn_g[4]  = {(const float*)d_in[5],  (const float*)d_in[9],
                            (const float*)d_in[13], (const float*)d_in[17]};
  const float* ebn_b[4]  = {(const float*)d_in[6],  (const float*)d_in[10],
                            (const float*)d_in[14], (const float*)d_in[18]};
  const float* fc5_w = (const float*)d_in[19]; const float* fc5_b = (const float*)d_in[20];
  const float* fc6_w = (const float*)d_in[21]; const float* fc6_b = (const float*)d_in[22];
  const float* dcw[4]   = {(const float*)d_in[23], (const float*)d_in[26],
                           (const float*)d_in[29], (const float*)d_in[32]};
  const float* dbn_g[4] = {(const float*)d_in[24], (const float*)d_in[27],
                           (const float*)d_in[30], (const float*)d_in[33]};
  const float* dbn_b[4] = {(const float*)d_in[25], (const float*)d_in[28],
                           (const float*)d_in[31], (const float*)d_in[34]};
  const float* mlp_w[4] = {(const float*)d_in[35], (const float*)d_in[37],
                           (const float*)d_in[39], (const float*)d_in[41]};
  const float* mlp_b[4] = {(const float*)d_in[36], (const float*)d_in[38],
                           (const float*)d_in[40], (const float*)d_in[42]};

  // bump allocator over workspace (256B-aligned -> all GEMM operands 16B OK)
  char* cur = (char*)d_ws;
  auto allocB = [&](size_t bytes) { char* p = cur; cur += (bytes + 255) & ~(size_t)255; return p; };
#define ALLOC_H(n) (h16*)allocB((size_t)(n) * sizeof(h16))
#define ALLOC_F(n) (float*)allocB((size_t)(n) * sizeof(float))

  float* grid32 = ALLOC_F((long)Bn * 262144);
  h16*   x64h   = ALLOC_H((long)Bn * 262144);

  const int eCi[4] = {1, 32, 64, 128};
  const int eCo[4] = {32, 64, 128, 256};
  const int eD [4] = {64, 32, 16, 8};
  h16* wch[4];
  for (int s = 0; s < 4; ++s) wch[s] = ALLOC_H((long)eCo[s] * eCi[s] * 64);
  h16* fc5wh = ALLOC_H((long)2048 * 16384);
  h16* fc6wh = ALLOC_H((long)16384 * 2048);
  const int mlpM[4] = {512, 256, 128, 1};
  const int mlpK[4] = {16, 512, 256, 128};
  h16* mlpwh[4];
  for (int s = 0; s < 4; ++s) mlpwh[s] = ALLOC_H((long)mlpM[s] * mlpK[s]);
  const int dCi[4] = {256, 128, 64, 32};
  const int dCo[4] = {128, 64, 32, 16};
  const int dDi[4] = {4, 8, 16, 32};
  h16* dwp[4];
  for (int s = 0; s < 4; ++s) dwp[s] = ALLOC_H((long)8 * dCo[s] * dCi[s] * 8);

  const long NCHUNK = 4096;                       // multiple of 8
  h16* colbuf = ALLOC_H((long)8192 * NCHUNK);     // 64 MB, also fits 256x131072

  h16* cbuf[4]; h16* pbuf[4];
  long eNfull[4], ePcnt[4];
  for (int s = 0; s < 4; ++s) {
    long O = eD[s] + 1, P = eD[s] / 2;
    eNfull[s] = (long)Bn * O * O * O;
    ePcnt[s]  = (long)eCo[s] * Bn * P * P * P;
    cbuf[s] = ALLOC_H((long)eCo[s] * eNfull[s]);
    pbuf[s] = ALLOC_H(ePcnt[s]);
  }
  float* bnMean = ALLOC_F(256);
  float* bnIstd = ALLOC_F(256);
  h16* xfc = ALLOC_H((long)16384 * 8);   // N padded 4 -> 8
  h16* h5  = ALLOC_H((long)2048 * 8);
  h16* h6  = ALLOC_H((long)16384 * 8);
  h16* f4r = ALLOC_H((long)256 * Bn * 64);
  h16* decbuf = ALLOC_H((long)16 * Bn * 32768);   // max class output (stage 10)
  h16* dfull[4]; h16* drelu[4];
  for (int s = 0; s < 4; ++s) {
    long O = 2L * dDi[s];
    long sz = (long)dCo[s] * Bn * O * O * O;
    dfull[s] = ALLOC_H(sz);
    drelu[s] = ALLOC_H(sz);
  }
  h16* feat = ALLOC_H((long)16 * Bn * NQ);
  h16* ybuf[3];
  for (int s = 0; s < 3; ++s) ybuf[s] = ALLOC_H((long)mlpM[s] * Bn * NQ);

  auto gemm = [&](const h16* A, const h16* Bm, const float* bias, float* Cf, h16* Ch,
                  int M, int N, int K, int ldc, int nstore, int act) {
    dim3 g((unsigned)((N + 63) / 64), (unsigned)((M + 63) / 64));
    k_gemm<<<g, 256, 0, stream>>>(A, Bm, bias, Cf, Ch, M, N, K, ldc, nstore, act);
  };

  // ---- gridding ----
  k_zero_f32<<<nblk((long)Bn * 262144), 256, 0, stream>>>(grid32, (long)Bn * 262144);
  k_gridding<<<(Bn * NP + 255) / 256, 256, 0, stream>>>(partial, grid32, Bn, NP);
  k_f32_to_f16<<<nblk((long)Bn * 262144), 256, 0, stream>>>(grid32, x64h, (long)Bn * 262144);

  // ---- weight conversion / packing ----
  for (int s = 0; s < 4; ++s) {
    long n = (long)eCo[s] * eCi[s] * 64;
    k_f32_to_f16<<<nblk(n), 256, 0, stream>>>(conv_w[s], wch[s], n);
  }
  k_f32_to_f16<<<4096, 256, 0, stream>>>(fc5_w, fc5wh, (long)2048 * 16384);
  k_f32_to_f16<<<4096, 256, 0, stream>>>(fc6_w, fc6wh, (long)16384 * 2048);
  for (int s = 0; s < 4; ++s) {
    long n = (long)mlpM[s] * mlpK[s];
    k_f32_to_f16<<<nblk(n), 256, 0, stream>>>(mlp_w[s], mlpwh[s], n);
    long nd = (long)8 * dCo[s] * dCi[s] * 8;
    k_pack_deconv_w<<<nblk(nd), 256, 0, stream>>>(dcw[s], dwp[s], dCi[s], dCo[s]);
  }

  // ---- encoder: conv(im2col + WMMA GEMM) -> BN stats -> BN+lrelu+pool ----
  const h16* ein = x64h;
  for (int s = 0; s < 4; ++s) {
    int Ci = eCi[s], Co = eCo[s], D = eD[s], O = D + 1, P = D / 2;
    int K = Ci * 64;
    long Nfull = eNfull[s];
    for (long c0 = 0; c0 < Nfull; c0 += NCHUNK) {
      long rem = Nfull - c0;
      int ncAct = (int)(rem < NCHUNK ? rem : NCHUNK);
      int ncPad = (ncAct + 7) & ~7;               // keep N % 8 == 0
      long tot = (long)K * ncPad;
      k_im2col_conv<<<nblk(tot), 256, 0, stream>>>(ein, colbuf, Ci, Bn, D, O,
                                                   c0, ncPad, Nfull);
      gemm(wch[s], colbuf, conv_b[s], nullptr, cbuf[s] + c0,
           Co, ncPad, K, (int)Nfull, ncAct, 0);
    }
    k_bn_stats<<<Co, 256, 0, stream>>>(cbuf[s], (int)Nfull, bnMean, bnIstd, 1e-5f);
    k_bn_lrelu_pool<<<nblk(ePcnt[s]), 256, 0, stream>>>(cbuf[s], bnMean, bnIstd,
                                                        ebn_g[s], ebn_b[s], pbuf[s],
                                                        Co, Bn, O, P);
    ein = pbuf[s];
  }

  // ---- bottleneck FC (WMMA) ----
  k_pack_fc<<<nblk((long)16384 * 8), 256, 0, stream>>>(pbuf[3], xfc, Bn);
  gemm(fc5wh, xfc, fc5_b, nullptr, h5, 2048, 8, 16384, 8, 8, 1);
  gemm(fc6wh, h5,  fc6_b, nullptr, h6, 16384, 8, 2048, 8, 8, 1);
  k_add_fc<<<nblk((long)256 * Bn * 64), 256, 0, stream>>>(pbuf[3], h6, f4r, Bn);

  // ---- decoder: 8-parity deconv GEMMs -> scatter -> BN+relu (+skip) ----
  const h16* din = f4r;
  const h16* skips[4] = {pbuf[2], pbuf[1], pbuf[0], nullptr};
  for (int s = 0; s < 4; ++s) {
    int Ci = dCi[s], Co = dCo[s], D = dDi[s], O = 2 * D;
    int K = Ci * 8;
    long Ncols = (long)Bn * D * D * D;            // multiple of 8
    for (int par = 0; par < 8; ++par) {
      long tot = (long)K * Ncols;
      k_im2col_deconv<<<nblk(tot), 256, 0, stream>>>(din, colbuf, Ci, Bn, D, par);
      gemm(dwp[s] + (size_t)par * Co * K, colbuf, nullptr, nullptr, decbuf,
           Co, (int)Ncols, K, (int)Ncols, (int)Ncols, 0);
      k_scatter_par<<<nblk((long)Co * Ncols), 256, 0, stream>>>(decbuf, dfull[s],
                                                                Co, Bn, D, par);
    }
    long L = (long)Bn * O * O * O;
    k_bn_stats<<<Co, 256, 0, stream>>>(dfull[s], (int)L, bnMean, bnIstd, 1e-5f);
    k_bn_relu_add<<<nblk((long)Co * L), 256, 0, stream>>>(dfull[s], bnMean, bnIstd,
                                                          dbn_g[s], dbn_b[s],
                                                          skips[s], drelu[s], Co, L);
    din = drelu[s];
  }

  // ---- projection + pointwise MLP (WMMA) + sigmoid + MSE ----
  k_project<<<nblk((long)16 * Bn * NQ), 256, 0, stream>>>(points, drelu[3], feat, Bn, NQ);
  const int NPTS = Bn * NQ;   // 32768, multiple of 8
  gemm(mlpwh[0], feat,    mlp_b[0], nullptr, ybuf[0], 512, NPTS, 16,  NPTS, NPTS, 2);
  gemm(mlpwh[1], ybuf[0], mlp_b[1], nullptr, ybuf[1], 256, NPTS, 512, NPTS, NPTS, 2);
  gemm(mlpwh[2], ybuf[1], mlp_b[2], nullptr, ybuf[2], 128, NPTS, 256, NPTS, NPTS, 2);
  float* preds = (float*)d_out;
  gemm(mlpwh[3], ybuf[2], mlp_b[3], preds, nullptr, 1, NPTS, 128, NPTS, NPTS, 3);
  k_zero_f32<<<1, 1, 0, stream>>>(preds + NPTS, 1);
  k_mse<<<64, 256, 0, stream>>>(preds, labels, preds + NPTS, NPTS);
#undef ALLOC_H
#undef ALLOC_F
}
